// GlobalPoolDistance_34935263986419
// MI455X (gfx1250) — matrix operand order repaired and checked by reading, not compiled, not run on hardware
//
#include <hip/hip_runtime.h>
#include <hip/hip_bf16.h>

typedef __attribute__((ext_vector_type(16))) _Float16 v16h;
typedef __attribute__((ext_vector_type(8)))  float    v8f;

#define BATCH   8
#define CH      3
#define IMG     64
#define PW      62            // patches per row/col (64-2)
#define NPATCH  3844          // 62*62
#define NT      241           // ceil(3844/16) row/col tiles
#define NPAD    (NT*16)       // 3856
#define KF      27            // 3*3*3 feature dim, padded to 32 for WMMA
#define NWAVE   (3*BATCH*NT)  // 5784 waves total

// exp(-d2/sig2) = exp2(-d2 * CSCALE), sig2 = (0.06*9)^2 = 0.2916
#define CSCALE  (1.4426950408889634f / 0.2916f)   // log2(e)/sig2
#define TWOC    (2.0f * CSCALE)

// ---------------------------------------------------------------------------
// Stage 1: pack patch features straight into WMMA fragment layouts (f16).
//   A-fragment (16x32, MxK): lane m = l&15; hi = l>>4;
//       e<8  -> k = hi*8 + e ; e>=8 -> k = 16 + hi*8 + (e-8)
//   B-fragment (32x16, KxN): lane n = l&15; same k mapping as A
// k in [27,32) and rows >= 3844 are zero-filled (contribute 0 to the Gram).
// ---------------------------------------------------------------------------
__global__ void pack_frags(const float* __restrict__ img,
                           _Float16* __restrict__ outA,
                           _Float16* __restrict__ outB) {
    int tid = blockIdx.x * blockDim.x + threadIdx.x;
    const int total = BATCH * NT * 32;
    if (tid >= total) return;
    int lane = tid & 31;
    int t    = (tid >> 5) % NT;
    int b    = tid / (NT * 32);
    int mn   = lane & 15;
    int g    = t * 16 + mn;          // global patch index (row for A, col for B)
    int hi   = lane >> 4;
    bool gok = g < NPATCH;
    int py = g / PW, px = g % PW;

    v16h va, vb;
#pragma unroll
    for (int e = 0; e < 16; ++e) {
        int ka = (e < 8) ? (hi * 8 + e) : (16 + hi * 8 + (e - 8));
        int kb = (e < 8) ? (hi * 8 + e) : (16 + hi * 8 + (e - 8));
        float fa = 0.f, fb = 0.f;
        if (gok && ka < KF) {
            int c = ka / 9, rem = ka - c * 9, r = rem / 3, s = rem - r * 3;
            fa = img[((b * CH + c) * IMG + py + r) * IMG + px + s];
        }
        if (gok && kb < KF) {
            int c = kb / 9, rem = kb - c * 9, r = rem / 3, s = rem - r * 3;
            fb = img[((b * CH + c) * IMG + py + r) * IMG + px + s];
        }
        va[e] = (_Float16)fa;
        vb[e] = (_Float16)fb;
    }
    *(v16h*)(outA + (size_t)tid * 16) = va;   // 32B aligned vector store
    *(v16h*)(outB + (size_t)tid * 16) = vb;
}

// ---------------------------------------------------------------------------
// Stage 2: squared norms of the f16-rounded features, pre-scaled by
// log2(e)/sig2 so the GEMM epilogue is a single pk_add + pk_fma + v_exp_f32.
// Padded patch slots get +INF: exp2(finite - INF) == 0 -> free masking.
// ---------------------------------------------------------------------------
__global__ void sqnorms(const float* __restrict__ img, float* __restrict__ sq) {
    int tid = blockIdx.x * blockDim.x + threadIdx.x;
    const int total = BATCH * NPAD;
    if (tid >= total) return;
    int g = tid % NPAD;
    int b = tid / NPAD;
    float out = __builtin_inff();
    if (g < NPATCH) {
        int py = g / PW, px = g % PW;
        float s = 0.f;
#pragma unroll
        for (int k = 0; k < KF; ++k) {
            int c = k / 9, rem = k - c * 9, r = rem / 3, ss = rem - r * 3;
            float v = img[((b * CH + c) * IMG + py + r) * IMG + px + ss];
            float h = (float)(_Float16)v;
            s += h * h;
        }
        out = s * CSCALE;
    }
    sq[tid] = out;
}

// One 16x16 output tile: WMMA + fused exp2 epilogue (no masks, no selects).
__device__ __forceinline__ void tile_accum(const v16h& a,
                                           const _Float16* __restrict__ bf_ptr,
                                           float sc, const v8f& srv,
                                           v8f& accsum) {
    v16h bf = *(const v16h*)bf_ptr;
    v8f c = {};
    v8f acc = __builtin_amdgcn_wmma_f32_16x16x32_f16(
        /*neg_a=*/false, a, /*neg_b=*/false, bf,
        /*c_mod=*/(short)0, c, /*reuse_a=*/false, /*reuse_b=*/false);
#pragma unroll
    for (int r = 0; r < 8; ++r) {
        float t   = srv[r] + sc;                        // pk_add
        float arg = __builtin_fmaf(acc[r], TWOC, -t);   // pk_fma
        accsum[r] += __builtin_amdgcn_exp2f(arg);       // v_exp_f32 (+pk_add)
    }
}

// ---------------------------------------------------------------------------
// Stage 3: one wave per (pair, batch, row-tile). 723 blocks * 256 thr =
// exactly 5784 waves -> every wave fully active (EXEC all-1s for WMMA).
// Wave id forced scalar via readfirstlane -> scalar loop control.
// XY waves sweep all 241 B tiles; XX/YY waves exploit Gram symmetry and
// sweep only tj >= ti; first tile peeled (weight 1), remaining tiles
// accumulated unweighted and scaled by 2 once at the end (exact: transpose
// tiles are bit-identical WMMA results).
// ---------------------------------------------------------------------------
__global__ void __launch_bounds__(256)
mmd_wmma(const _Float16* __restrict__ AX, const _Float16* __restrict__ BX,
         const _Float16* __restrict__ AY, const _Float16* __restrict__ BY,
         const float* __restrict__ sqX,  const float* __restrict__ sqY,
         float* __restrict__ partials) {
    int gtid = blockIdx.x * 256 + threadIdx.x;
    int wave = __builtin_amdgcn_readfirstlane(gtid >> 5);   // wave-uniform -> SGPR
    int lane = threadIdx.x & 31;
    if (wave >= NWAVE) return;             // never taken (exact fit), scalar guard

    int ti = wave % NT;
    int b  = (wave / NT) % BATCH;
    int p  = wave / (NT * BATCH);          // 0: XY (w=-2)  1: XX  2: YY

    const _Float16* Afrag = (p == 2) ? AY : AX;
    const _Float16* Bfrag = (p == 1) ? BX : BY;
    const float*    sqA   = (p == 2) ? sqY : sqX;
    const float*    sqB   = (p == 1) ? sqX : sqY;
    bool  sym   = (p != 0);
    float w     = sym ? 1.0f : -2.0f;
    float mbase = sym ? 2.0f : 1.0f;       // off-diagonal tile weight
    int   tj0   = sym ? ti : 0;            // scalar

    v16h a = *(const v16h*)(Afrag + ((size_t)(b * NT + ti) * 32 + lane) * 16);

    // This lane's 8 output rows (consecutive, 32B aligned): one v8f load.
    int rbase = ti * 16 + ((lane >> 4) << 3);
    v8f srv = *(const v8f*)(sqA + b * NPAD + rbase);

    int ncol = lane & 15;
    const _Float16* bft = Bfrag + ((size_t)(b * NT + tj0) * 32 + lane) * 16;
    const float*    scp = sqB + b * NPAD + tj0 * 16 + ncol;

    // Peeled first tile: weight 1 for every pair type.
    v8f accDiag = {}, accMain = {};
    tile_accum(a, bft, *scp, srv, accDiag);

    for (int tj = tj0 + 1; tj < NT; ++tj) {
        bft += 512;                        // next B fragment (running pointers)
        scp += 16;
        __builtin_prefetch(bft + 512, 0, 3);   // global_prefetch_b8 (speculative)
        tile_accum(a, bft, *scp, srv, accMain);
    }

    float sd = 0.f, sm = 0.f;
#pragma unroll
    for (int r = 0; r < 8; ++r) { sd += accDiag[r]; sm += accMain[r]; }
    float sum = __builtin_fmaf(mbase, sm, sd);

    // wave32 butterfly reduction
#pragma unroll
    for (int off = 16; off > 0; off >>= 1)
        sum += __shfl_xor(sum, off, 32);
    if (lane == 0) partials[wave] = w * sum;   // fixed slot -> deterministic
}

// ---------------------------------------------------------------------------
// Stage 4: deterministic fixed-order final reduction (no float atomics).
// ---------------------------------------------------------------------------
__global__ void final_reduce(const float* __restrict__ partials,
                             float* __restrict__ out) {
    __shared__ float smem[256];
    float s = 0.f;
    for (int i = threadIdx.x; i < NWAVE; i += 256) s += partials[i];
    smem[threadIdx.x] = s;
    __syncthreads();
#pragma unroll
    for (int step = 128; step > 0; step >>= 1) {
        if ((int)threadIdx.x < step) smem[threadIdx.x] += smem[threadIdx.x + step];
        __syncthreads();
    }
    if (threadIdx.x == 0)
        out[0] = smem[0] * (1.0f / ((float)BATCH * (float)NPATCH * (float)NPATCH));
}

extern "C" void kernel_launch(void* const* d_in, const int* in_sizes, int n_in,
                              void* d_out, int out_size, void* d_ws, size_t ws_size,
                              hipStream_t stream) {
    const float* x = (const float*)d_in[0];
    const float* y = (const float*)d_in[1];
    float* out = (float*)d_out;

    // Workspace layout (~8.2 MB, fully L2-resident on 192MB L2)
    const size_t FE = (size_t)BATCH * NT * 512;   // halves per fragment buffer
    _Float16* AX = (_Float16*)d_ws;
    _Float16* BX = AX + FE;
    _Float16* AY = BX + FE;
    _Float16* BY = AY + FE;
    float* sqX = (float*)(BY + FE);
    float* sqY = sqX + BATCH * NPAD;
    float* partials = sqY + BATCH * NPAD;

    const int packThreads = BATCH * NT * 32;      // 61696
    pack_frags<<<(packThreads + 255) / 256, 256, 0, stream>>>(x, AX, BX);
    pack_frags<<<(packThreads + 255) / 256, 256, 0, stream>>>(y, AY, BY);

    const int sqThreads = BATCH * NPAD;           // 30848
    sqnorms<<<(sqThreads + 255) / 256, 256, 0, stream>>>(x, sqX);
    sqnorms<<<(sqThreads + 255) / 256, 256, 0, stream>>>(y, sqY);

    mmd_wmma<<<(NWAVE * 32) / 256, 256, 0, stream>>>(AX, BX, AY, BY,
                                                     sqX, sqY, partials);
    final_reduce<<<1, 256, 0, stream>>>(partials, out);
}
